// Decoder_41420664603083
// MI455X (gfx1250) — compile-verified, matrix operand order checked by reading, not run on hardware
//
#include <hip/hip_runtime.h>
#include <hip/hip_bf16.h>

typedef __attribute__((ext_vector_type(2)))  _Float16 v2h;
typedef __attribute__((ext_vector_type(4)))  _Float16 v4h;
typedef __attribute__((ext_vector_type(8)))  _Float16 v8h;
typedef __attribute__((ext_vector_type(16))) _Float16 v16h;
typedef __attribute__((ext_vector_type(4)))  float    v4f;
typedef __attribute__((ext_vector_type(8)))  float    v8f;

#define N_LAYERS 3
#define HN   8
#define DKH  64
#define DM   768
#define DFF  2048
#define VOC  32000
#define BBV  8
#define LLV  256
#define ENCL 257
#define LKP  320   // encoder seq padded to multiple of 64 (and 32 for K-loop)
#define HD   512   // HN*DKH

// pack 4 f32 -> 4 f16 with two v_cvt_pk instructions
__device__ inline v4h cvt4(v4f v)
{
  v2h lo = __builtin_bit_cast(v2h, __builtin_amdgcn_cvt_pkrtz(v[0], v[1]));
  v2h hi = __builtin_bit_cast(v2h, __builtin_amdgcn_cvt_pkrtz(v[2], v[3]));
  return __builtin_shufflevector(lo, hi, 0, 1, 2, 3);
}

__device__ inline v16h frag16(const _Float16* p, int halfw)
{
  v8h lo = *(const v8h*)(p + (halfw << 3));
  v8h hi = *(const v8h*)(p + 16 + (halfw << 3));
  return __builtin_shufflevector(lo, hi, 0, 1, 2, 3, 4, 5, 6, 7,
                                 8, 9, 10, 11, 12, 13, 14, 15);
}

// ---------------------------------------------------------------------------
// BIG GEMM: 256 threads = 8 waves, 128x128 block tile (M,N multiples of 128).
// Wave (wv&3) -> 32-row M band (2 frags), (wv>>2) -> 64-col N half (4 subs):
// 8 WMMAs per K-step, A-frag reuse x4, B-frag reuse x2. 16 KB LDS.
// A: M-major contiguous K. B: N-contiguous row-major weights (k*sB + n).
// FLAGS: 1=+bias[n], 2=relu, 4=+resid (C layout)
// ---------------------------------------------------------------------------
template<int FLAGS>
__global__ void __launch_bounds__(256)
gemm_wmma_big(const float* __restrict__ A, long sAM,
              const float* __restrict__ Bm, long sB,
              float* __restrict__ C, long sCM,
              const float* __restrict__ bias, const float* __restrict__ resid,
              int K, float scale)
{
  const int tid  = threadIdx.x;
  const int lane = tid & 31;
  const int wv   = tid >> 5;             // 0..7
  const int mw   = (wv & 3) * 32;        // M band within tile
  const int nh   = (wv >> 2) * 64;       // N half within tile
  const int m0 = blockIdx.x * 128;
  const int n0 = blockIdx.y * 128;

  __shared__ _Float16 As[128 * 32];      // [m][k]
  __shared__ _Float16 Bs[128 * 32];      // [n][k] (N-major)

  // A staging: r = q*32 + tid>>3, c4 = (tid&7)*4
  const float* pAk = A + (long)(m0 + (tid >> 3)) * sAM + ((tid & 7) << 2);
  const long   sAq = sAM * 32;
  _Float16* sAp = &As[(tid >> 3) * 32 + ((tid & 7) << 2)];
  // B staging: kr = q*8 + tid>>5, c4 = (tid&31)*4
  const float* pBk = Bm + (long)(tid >> 5) * sB + n0 + ((tid & 31) << 2);
  const long   sBq = sB * 8;
  _Float16* sBp = &Bs[((tid & 31) << 2) * 32 + (tid >> 5)];

  v8f acc[2][4] = {};
  const int mloc = lane & 15, halfw = lane >> 4;

  for (int k0 = 0; k0 < K; k0 += 32) {
    // phase 1: clause of global loads
    v4f ar[4], br[4];
    #pragma unroll
    for (int q = 0; q < 4; ++q) ar[q] = *(const v4f*)(pAk + q * sAq);
    #pragma unroll
    for (int q = 0; q < 4; ++q) br[q] = *(const v4f*)(pBk + q * sBq);
    pAk += 32;
    pBk += (long)32 * sB;
    // phase 2: packed cvt + LDS
    #pragma unroll
    for (int q = 0; q < 4; ++q) *(v4h*)(sAp + q * 1024) = cvt4(ar[q]);
    #pragma unroll
    for (int q = 0; q < 4; ++q) {        // scatter across N-major rows
      v4h hv = cvt4(br[q]);
      sBp[q * 8]      = hv[0];
      sBp[q * 8 + 32] = hv[1];
      sBp[q * 8 + 64] = hv[2];
      sBp[q * 8 + 96] = hv[3];
    }
    __syncthreads();

    v16h af[2];
    #pragma unroll
    for (int mf = 0; mf < 2; ++mf)
      af[mf] = frag16(As + (mw + mf * 16 + mloc) * 32, halfw);
    #pragma unroll
    for (int sub = 0; sub < 4; ++sub) {
      v16h bf = frag16(Bs + (nh + sub * 16 + mloc) * 32, halfw);
      #pragma unroll
      for (int mf = 0; mf < 2; ++mf)
        acc[mf][sub] = __builtin_amdgcn_wmma_f32_16x16x32_f16(
            false, af[mf], false, bf, (short)0, acc[mf][sub], false, false);
    }
    __syncthreads();
  }

  #pragma unroll
  for (int mf = 0; mf < 2; ++mf) {
    #pragma unroll
    for (int sub = 0; sub < 4; ++sub) {
      int gn = n0 + nh + sub * 16 + mloc;
      #pragma unroll
      for (int r = 0; r < 8; ++r) {
        int gm = m0 + mw + mf * 16 + r + 8 * halfw;
        float val = acc[mf][sub][r] * scale;
        if (FLAGS & 1) val += bias[gn];
        if (FLAGS & 2) val = fmaxf(val, 0.f);
        if (FLAGS & 4) val += resid[(long)gm * sCM + gn];
        C[(long)gm * sCM + gn] = val;
      }
    }
  }
}

// ---------------------------------------------------------------------------
// SMALL batched GEMM for attention: 128 threads = 4 waves, 64x64 tile.
// B tile shared across waves; each wave stages its 16x32 A band, 4 WMMAs/step.
// A: M-major contiguous K. B: BKC=true -> n*sB + k ; else k*sB + n.
// FLAGS: 8=causal(n>m) + scale, 16=n>=NK mask + scale, 0=plain
// ---------------------------------------------------------------------------
template<int FLAGS, bool BKC>
__global__ void __launch_bounds__(128)
gemm_wmma(const float* __restrict__ A, long sAM, long sAb, long sAh,
          const float* __restrict__ Bm, long sB, long sBb, long sBh,
          float* __restrict__ C, long sCM, long sCb, long sCh,
          int K, int Hn, float scale, int NK)
{
  const int tid  = threadIdx.x;
  const int lane = tid & 31;
  const int wv   = tid >> 5;                 // wave 0..3 -> M band
  const int m0 = blockIdx.x * 64 + wv * 16;
  const int n0 = blockIdx.y * 64;
  const int bh = blockIdx.z;
  const int b = bh / Hn, h = bh % Hn;

  const float* Ab = A  + (long)b * sAb + (long)h * sAh;
  const float* Bb = Bm + (long)b * sBb + (long)h * sBh;
  float*       Cb = C  + (long)b * sCb + (long)h * sCh;

  __shared__ _Float16 As[4][16 * 32];   // per-wave [m][k]
  __shared__ _Float16 Bs[64 * 32];      // [n][k] (N-major)

  const float* pAk = Ab + (long)(m0 + (lane >> 3)) * sAM + ((lane & 7) << 2);
  const long   sAq = sAM * 4;
  _Float16* sAp = &As[wv][(lane >> 3) * 32 + ((lane & 7) << 2)];
  const float* pBk;
  long sBq;
  _Float16* sBp;
  if (BKC) {        // n = q*16 + tid>>3, c4 = (tid&7)*4
    pBk = Bb + (long)(n0 + (tid >> 3)) * sB + ((tid & 7) << 2);
    sBq = sB * 16;
    sBp = &Bs[(tid >> 3) * 32 + ((tid & 7) << 2)];
  } else {          // kr = q*8 + tid>>4, c4 = (tid&15)*4
    pBk = Bb + (long)(tid >> 4) * sB + n0 + ((tid & 15) << 2);
    sBq = sB * 8;
    sBp = &Bs[((tid & 15) << 2) * 32 + (tid >> 4)];
  }
  const long sBstep = BKC ? 32 : 32 * sB;

  v8f acc[4] = {};
  const int mloc  = lane & 15;
  const int halfw = lane >> 4;

  for (int k0 = 0; k0 < K; k0 += 32) {
    v4f ar[4], br[4];
    #pragma unroll
    for (int q = 0; q < 4; ++q) ar[q] = *(const v4f*)(pAk + q * sAq);
    #pragma unroll
    for (int q = 0; q < 4; ++q) br[q] = *(const v4f*)(pBk + q * sBq);
    pAk += 32;
    pBk += sBstep;
    #pragma unroll
    for (int q = 0; q < 4; ++q) *(v4h*)(sAp + q * 128) = cvt4(ar[q]);
    if (BKC) {
      #pragma unroll
      for (int q = 0; q < 4; ++q) *(v4h*)(sBp + q * 512) = cvt4(br[q]);
    } else {
      #pragma unroll
      for (int q = 0; q < 4; ++q) {
        v4h hv = cvt4(br[q]);
        sBp[q * 8]      = hv[0];
        sBp[q * 8 + 32] = hv[1];
        sBp[q * 8 + 64] = hv[2];
        sBp[q * 8 + 96] = hv[3];
      }
    }
    __syncthreads();

    v16h af = frag16(&As[wv][mloc * 32], halfw);
    #pragma unroll
    for (int sub = 0; sub < 4; ++sub) {
      v16h bf = frag16(Bs + (sub * 16 + mloc) * 32, halfw);
      acc[sub] = __builtin_amdgcn_wmma_f32_16x16x32_f16(false, af, false, bf,
                                                        (short)0, acc[sub],
                                                        false, false);
    }
    __syncthreads();
  }

  #pragma unroll
  for (int sub = 0; sub < 4; ++sub) {
    int gn = n0 + sub * 16 + mloc;
    #pragma unroll
    for (int r = 0; r < 8; ++r) {
      int gm = m0 + r + 8 * halfw;
      float val = acc[sub][r] * scale;
      if (FLAGS & 8)  { if (gn > gm)  val = -1e30f; }
      if (FLAGS & 16) { if (gn >= NK) val = -1e30f; }
      Cb[(long)gm * sCM + gn] = val;
    }
  }
}

// ---------------------------------------------------------------------------
// x[tok,:] = word2vector[ids[tok],:] + sinusoid_pe(l+1, :)
// ---------------------------------------------------------------------------
__global__ void embed_kernel(const int* __restrict__ ids,
                             const float* __restrict__ w2v,
                             float* __restrict__ x)
{
  int tok = blockIdx.x;             // 0 .. B*L-1
  int l   = tok % LLV;
  int id  = ids[tok];
  float pos = (float)(l + 1);       // no padding tokens -> seq = 1..L
  for (int j = threadIdx.x; j < DM; j += blockDim.x) {
    int jj = j & ~1;                // 2*(j//2)
    float ang = pos * powf(10000.f, -(float)jj / (float)DM);
    float pe  = (j & 1) ? cosf(ang) : sinf(ang);
    x[(long)tok * DM + j] = w2v[(long)id * DM + j] + pe;
  }
}

// ---------------------------------------------------------------------------
// encP[b, 0:320, :] = { enc[b, l, :] for l<257 ; 0 otherwise }
// ---------------------------------------------------------------------------
__global__ void pad_encoder(const float* __restrict__ enc,
                            float* __restrict__ encp)
{
  int row = blockIdx.x;             // 0 .. B*LKP-1
  int b = row / LKP, l = row % LKP;
  float* o = encp + (long)row * DM;
  if (l < ENCL) {
    const float* s = enc + ((long)b * ENCL + l) * DM;
    for (int j = threadIdx.x; j < DM; j += blockDim.x) o[j] = s[j];
  } else {
    for (int j = threadIdx.x; j < DM; j += blockDim.x) o[j] = 0.f;
  }
}

// ---------------------------------------------------------------------------
__global__ void layernorm_rows(float* __restrict__ x,
                               const float* __restrict__ w,
                               const float* __restrict__ bb, int D)
{
  long row = blockIdx.x;
  float* p = x + row * (long)D;
  __shared__ float r1[256], r2[256];
  int t = threadIdx.x;
  float s = 0.f, s2 = 0.f;
  for (int j = t; j < D; j += 256) { float v = p[j]; s += v; s2 += v * v; }
  r1[t] = s; r2[t] = s2; __syncthreads();
  for (int o = 128; o > 0; o >>= 1) {
    if (t < o) { r1[t] += r1[t + o]; r2[t] += r2[t + o]; }
    __syncthreads();
  }
  float mu   = r1[0] / (float)D;
  float var  = r2[0] / (float)D - mu * mu;
  float rstd = rsqrtf(var + 1e-5f);
  for (int j = t; j < D; j += 256) p[j] = (p[j] - mu) * rstd * w[j] + bb[j];
}

// ---------------------------------------------------------------------------
__global__ void softmax_rows(float* __restrict__ sc, int cols)
{
  long row = blockIdx.x;
  float* p = sc + row * (long)cols;
  __shared__ float red[256];
  int t = threadIdx.x;
  float m = -3.4e38f;
  for (int j = t; j < cols; j += 256) m = fmaxf(m, p[j]);
  red[t] = m; __syncthreads();
  for (int o = 128; o > 0; o >>= 1) {
    if (t < o) red[t] = fmaxf(red[t], red[t + o]);
    __syncthreads();
  }
  float mx = red[0];
  __syncthreads();
  float s = 0.f;
  for (int j = t; j < cols; j += 256) { float e = expf(p[j] - mx); p[j] = e; s += e; }
  red[t] = s; __syncthreads();
  for (int o = 128; o > 0; o >>= 1) {
    if (t < o) red[t] += red[t + o];
    __syncthreads();
  }
  float inv = 1.f / red[0];
  for (int j = t; j < cols; j += 256) p[j] *= inv;
}

// ---------------------------------------------------------------------------
__global__ void log_softmax_rows(const float* __restrict__ lg,
                                 float* __restrict__ o, int cols)
{
  long row = blockIdx.x;
  const float* p = lg + row * (long)cols;
  float* q = o + row * (long)cols;
  __shared__ float red[256];
  int t = threadIdx.x;
  float m = -3.4e38f;
  for (int j = t; j < cols; j += 256) m = fmaxf(m, p[j]);
  red[t] = m; __syncthreads();
  for (int o2 = 128; o2 > 0; o2 >>= 1) {
    if (t < o2) red[t] = fmaxf(red[t], red[t + o2]);
    __syncthreads();
  }
  float mx = red[0];
  __syncthreads();
  float s = 0.f;
  for (int j = t; j < cols; j += 256) s += expf(p[j] - mx);
  red[t] = s; __syncthreads();
  for (int o2 = 128; o2 > 0; o2 >>= 1) {
    if (t < o2) red[t] += red[t + o2];
    __syncthreads();
  }
  float lse = mx + logf(red[0]);
  for (int j = t; j < cols; j += 256) q[j] = p[j] - lse;
}

// ---------------------------------------------------------------------------
static inline dim3 gg64(int M, int N, int batch)
{ return dim3((unsigned)(M / 64), (unsigned)(N / 64), (unsigned)batch); }
static inline dim3 gg128(int M, int N)
{ return dim3((unsigned)(M / 128), (unsigned)(N / 128), 1u); }

extern "C" void kernel_launch(void* const* d_in, const int* in_sizes, int n_in,
                              void* d_out, int out_size, void* d_ws, size_t ws_size,
                              hipStream_t stream)
{
  (void)in_sizes; (void)n_in; (void)out_size; (void)ws_size;

  const int*   ids   = (const int*)  d_in[0];
  const float* enc   = (const float*)d_in[1];
  const float* w2v   = (const float*)d_in[2];
  const float* sa_Wq = (const float*)d_in[3];
  const float* sa_bq = (const float*)d_in[4];
  const float* sa_Wk = (const float*)d_in[5];
  const float* sa_bk = (const float*)d_in[6];
  const float* sa_Wv = (const float*)d_in[7];
  const float* sa_bv = (const float*)d_in[8];
  const float* sa_Wo = (const float*)d_in[9];
  const float* sa_bo = (const float*)d_in[10];
  const float* sa_lw = (const float*)d_in[11];
  const float* sa_lb = (const float*)d_in[12];
  const float* ea_Wq = (const float*)d_in[13];
  const float* ea_bq = (const float*)d_in[14];
  const float* ea_Wk = (const float*)d_in[15];
  const float* ea_bk = (const float*)d_in[16];
  const float* ea_Wv = (const float*)d_in[17];
  const float* ea_bv = (const float*)d_in[18];
  const float* ea_Wo = (const float*)d_in[19];
  const float* ea_bo = (const float*)d_in[20];
  const float* ea_lw = (const float*)d_in[21];
  const float* ea_lb = (const float*)d_in[22];
  const float* ff_W1 = (const float*)d_in[23];
  const float* ff_b1 = (const float*)d_in[24];
  const float* ff_W2 = (const float*)d_in[25];
  const float* ff_b2 = (const float*)d_in[26];
  const float* ff_lw = (const float*)d_in[27];
  const float* ff_lb = (const float*)d_in[28];
  const float* fc_W  = (const float*)d_in[29];

  float* out    = (float*)d_out;
  float* logits = out + (size_t)BBV * LLV * VOC;   // second half of d_out

  // Workspace carve-up (floats)
  float* ws = (float*)d_ws;
  size_t off = 0;
  float* X    = ws + off; off += (size_t)BBV * LLV * DM;       // ping
  float* Y    = ws + off; off += (size_t)BBV * LLV * DM;       // pong
  float* ENCP = ws + off; off += (size_t)BBV * LKP * DM;       // padded encoder
  float* Q    = ws + off; off += (size_t)BBV * LLV * HD;       // query proj
  float* KB   = ws + off; off += (size_t)BBV * LKP * HD;       // key proj
  float* VB   = ws + off; off += (size_t)BBV * LKP * HD;       // value proj
  float* S    = ws + off; off += (size_t)BBV * HN * LLV * LKP; // attn scores
  float* AO   = ws + off; off += (size_t)BBV * LLV * HD;       // attn output
  float* FH   = ws + off; off += (size_t)BBV * LLV * DFF;      // ffn hidden

  const int Mr  = BBV * LLV;       // 2048 decoder rows
  const int Mep = BBV * LKP;       // 2560 padded encoder rows
  const dim3 T128(128, 1, 1);
  const dim3 T256(256, 1, 1);

  embed_kernel<<<Mr, 256, 0, stream>>>(ids, w2v, X);
  pad_encoder<<<Mep, 256, 0, stream>>>(enc, ENCP);

  for (int i = 0; i < N_LAYERS; ++i) {
    const float *Wq = sa_Wq + (size_t)i * DM * HD, *bq = sa_bq + (size_t)i * HD;
    const float *Wk = sa_Wk + (size_t)i * DM * HD, *bk = sa_bk + (size_t)i * HD;
    const float *Wv = sa_Wv + (size_t)i * DM * HD, *bv = sa_bv + (size_t)i * HD;
    const float *Wo = sa_Wo + (size_t)i * HD * DM, *bo = sa_bo + (size_t)i * DM;
    const float *eWq = ea_Wq + (size_t)i * DM * HD, *ebq = ea_bq + (size_t)i * HD;
    const float *eWk = ea_Wk + (size_t)i * DM * HD, *ebk = ea_bk + (size_t)i * HD;
    const float *eWv = ea_Wv + (size_t)i * DM * HD, *ebv = ea_bv + (size_t)i * HD;
    const float *eWo = ea_Wo + (size_t)i * HD * DM, *ebo = ea_bo + (size_t)i * DM;
    const float *W1 = ff_W1 + (size_t)i * DM * DFF, *b1 = ff_b1 + (size_t)i * DFF;
    const float *W2 = ff_W2 + (size_t)i * DFF * DM, *b2 = ff_b2 + (size_t)i * DM;

    // ---- self-attention ----
    gemm_wmma_big<1><<<gg128(Mr, HD), T256, 0, stream>>>(X, DM, Wq, HD,
        Q, HD, bq, nullptr, DM, 1.f);
    gemm_wmma_big<1><<<gg128(Mr, HD), T256, 0, stream>>>(X, DM, Wk, HD,
        KB, HD, bk, nullptr, DM, 1.f);
    gemm_wmma_big<1><<<gg128(Mr, HD), T256, 0, stream>>>(X, DM, Wv, HD,
        VB, HD, bv, nullptr, DM, 1.f);
    // scores = causal( Q @ K^T / 8 ), batched over (b,h); B is K-contiguous
    gemm_wmma<8, true><<<gg64(LLV, LLV, BBV * HN), T128, 0, stream>>>(
        Q,  HD, (long)LLV * HD, DKH,
        KB, HD, (long)LLV * HD, DKH,
        S,  LLV, (long)HN * LLV * LLV, (long)LLV * LLV,
        DKH, HN, 0.125f, 0);
    softmax_rows<<<BBV * HN * LLV, 256, 0, stream>>>(S, LLV);
    // AO = P @ V
    gemm_wmma<0, false><<<gg64(LLV, DKH, BBV * HN), T128, 0, stream>>>(
        S,  LLV, (long)HN * LLV * LLV, (long)LLV * LLV,
        VB, HD, (long)LLV * HD, DKH,
        AO, HD, (long)LLV * HD, DKH,
        LLV, HN, 1.f, 0);
    // Y = LN( X + AO @ Wo + bo )
    gemm_wmma_big<5><<<gg128(Mr, DM), T256, 0, stream>>>(AO, HD, Wo, DM,
        Y, DM, bo, X, HD, 1.f);
    layernorm_rows<<<Mr, 256, 0, stream>>>(Y, sa_lw + (size_t)i * DM,
                                           sa_lb + (size_t)i * DM, DM);

    // ---- cross-attention (padded encoder keys, mask n>=257) ----
    gemm_wmma_big<1><<<gg128(Mr, HD), T256, 0, stream>>>(Y, DM, eWq, HD,
        Q, HD, ebq, nullptr, DM, 1.f);
    gemm_wmma_big<1><<<gg128(Mep, HD), T256, 0, stream>>>(ENCP, DM, eWk, HD,
        KB, HD, ebk, nullptr, DM, 1.f);
    gemm_wmma_big<1><<<gg128(Mep, HD), T256, 0, stream>>>(ENCP, DM, eWv, HD,
        VB, HD, ebv, nullptr, DM, 1.f);
    gemm_wmma<16, true><<<gg64(LLV, LKP, BBV * HN), T128, 0, stream>>>(
        Q,  HD, (long)LLV * HD, DKH,
        KB, HD, (long)LKP * HD, DKH,
        S,  LKP, (long)HN * LLV * LKP, (long)LLV * LKP,
        DKH, HN, 0.125f, ENCL);
    softmax_rows<<<BBV * HN * LLV, 256, 0, stream>>>(S, LKP);
    gemm_wmma<0, false><<<gg64(LLV, DKH, BBV * HN), T128, 0, stream>>>(
        S,  LKP, (long)HN * LLV * LKP, (long)LLV * LKP,
        VB, HD, (long)LKP * HD, DKH,
        AO, HD, (long)LLV * HD, DKH,
        LKP, HN, 1.f, 0);
    // X = LN( Y + AO @ eWo + ebo )
    gemm_wmma_big<5><<<gg128(Mr, DM), T256, 0, stream>>>(AO, HD, eWo, DM,
        X, DM, ebo, Y, HD, 1.f);
    layernorm_rows<<<Mr, 256, 0, stream>>>(X, ea_lw + (size_t)i * DM,
                                           ea_lb + (size_t)i * DM, DM);

    // ---- FFN ----
    gemm_wmma_big<3><<<gg128(Mr, DFF), T256, 0, stream>>>(X, DM, W1, DFF,
        FH, DFF, b1, nullptr, DM, 1.f);
    gemm_wmma_big<5><<<gg128(Mr, DM), T256, 0, stream>>>(FH, DFF, W2, DM,
        Y, DM, b2, X, DFF, 1.f);
    layernorm_rows<<<Mr, 256, 0, stream>>>(Y, ff_lw + (size_t)i * DM,
                                           ff_lb + (size_t)i * DM, DM);

    // ping-pong
    float* t = X; X = Y; Y = t;
  }

  // logits = X @ fc_W  (written straight into second half of d_out)
  gemm_wmma_big<0><<<gg128(Mr, VOC), T256, 0, stream>>>(X, DM, fc_W, VOC,
      logits, VOC, nullptr, nullptr, DM, 1.f);
  // first half of d_out = log_softmax(logits)
  log_softmax_rows<<<Mr, 256, 0, stream>>>(logits, out, VOC);
}
